// MambaBlock_20289425506777
// MI455X (gfx1250) — matrix operand order repaired and checked by reading, not compiled, not run on hardware
//
#include <hip/hip_runtime.h>

// ---------------------------------------------------------------------------
// Mamba block for MI455X (gfx1250): bf16 WMMA GEMMs (f32 accumulate),
// f32 recurrence. wave32 everywhere.
// ---------------------------------------------------------------------------

typedef __attribute__((ext_vector_type(16))) __bf16 v16bf;
typedef __attribute__((ext_vector_type(8)))  __bf16 v8bf;
typedef __attribute__((ext_vector_type(8)))  float  v8f;
typedef int v4i __attribute__((vector_size(16)));

#define D_MODEL 1024
#define D_INNER 2048
#define D_STATE 64
#define DT_RANK 64
#define NB      4
#define SEQ     2048
#define NTOK    (NB * SEQ)                 // 8192 tokens
#define XPROJ_N (DT_RANK + 2 * D_STATE)    // 192

#if defined(__AMDGCN__) && __has_builtin(__builtin_amdgcn_global_load_async_to_lds_b128)
#define HAVE_ASYNC_LDS 1
typedef __attribute__((address_space(1))) v4i gv4i;   // global b128 payload
typedef __attribute__((address_space(3))) v4i lv4i;   // LDS b128 payload
#else
#define HAVE_ASYNC_LDS 0
#endif

// ---------------------------- generic helpers ------------------------------

__global__ __launch_bounds__(256)
void cast_bf16_kernel(const float* __restrict__ in, __bf16* __restrict__ out, int n) {
  int i = blockIdx.x * 256 + threadIdx.x;
  if (i < n) out[i] = (__bf16)in[i];
}

__global__ __launch_bounds__(256)
void aneg_kernel(const float* __restrict__ a_log, float* __restrict__ an) {
  int i = blockIdx.x * 256 + threadIdx.x;   // D_INNER*D_STATE
  an[i] = -__expf(a_log[i]);
}

// ------------------------------- LayerNorm ---------------------------------
__global__ __launch_bounds__(256)
void ln_kernel(const float* __restrict__ x, const float* __restrict__ w,
               const float* __restrict__ b, __bf16* __restrict__ xn) {
  __shared__ float red[256];
  const int row = blockIdx.x;
  const int tid = threadIdx.x;
  const float* xr = x + (size_t)row * D_MODEL;

  float v[4];
  float s = 0.f;
#pragma unroll
  for (int j = 0; j < 4; ++j) { v[j] = xr[tid + j * 256]; s += v[j]; }
  red[tid] = s;
  __syncthreads();
  for (int off = 128; off > 0; off >>= 1) {
    if (tid < off) red[tid] += red[tid + off];
    __syncthreads();
  }
  const float mu = red[0] * (1.f / D_MODEL);
  __syncthreads();

  float sq = 0.f;
#pragma unroll
  for (int j = 0; j < 4; ++j) { float d = v[j] - mu; sq += d * d; }
  red[tid] = sq;
  __syncthreads();
  for (int off = 128; off > 0; off >>= 1) {
    if (tid < off) red[tid] += red[tid + off];
    __syncthreads();
  }
  const float rstd = rsqrtf(red[0] * (1.f / D_MODEL) + 1e-5f);

  __bf16* xo = xn + (size_t)row * D_MODEL;
#pragma unroll
  for (int j = 0; j < 4; ++j) {
    int c = tid + j * 256;
    xo[c] = (__bf16)(((v[j] - mu) * rstd) * w[c] + b[c]);
  }
}

// ----------------------------- WMMA GEMM -----------------------------------
// C[M,N] = A[M,K] * B[K,N] (+ residual), A/B bf16 row-major, C f32.
// Block tile 128x64, 8 waves in a 4x2 grid; each wave owns a 32x32 patch
// (2x2 WMMA tiles) -> 4 v_wmma per K-step with 2x fragment reuse.
#define BM 128
#define BN 64
#define BK 32
#define AS 40   // LDS row stride (bf16 elems): 80B, 16B-aligned, conflict pad
#define BS 40

__global__ __launch_bounds__(256)
void gemm_bf16_wmma(const __bf16* __restrict__ A, const __bf16* __restrict__ Bw,
                    float* __restrict__ C, const float* __restrict__ residual,
                    int M, int N, int K) {
  __shared__ __bf16 As[BM * AS];
  __shared__ __bf16 Bt[BN * BS];   // stored transposed: Bt[n][k]

  const int tid  = threadIdx.x;
  const int wave = tid >> 5;
  const int lane = tid & 31;
  const int m0 = blockIdx.y * BM;
  const int n0 = blockIdx.x * BN;

  const int wm = (wave >> 1) * 32;  // wave patch row (0,32,64,96)
  const int wn = (wave & 1) * 32;   // wave patch col (0,32)
  const int fr = lane & 15;
  const int hi = lane >> 4;         // lane half select

  // staging assignments
  const int arow = tid >> 2;        // 0..63 (+64 for second row)
  const int acol = (tid & 3) * 8;   // 0,8,16,24
  const int bk   = tid >> 3;        // 0..31
  const int bn   = (tid & 7) * 8;   // 0..56

  v8f acc[2][2] = {};

  const int nK = K / BK;
  for (int kt = 0; kt < nK; ++kt) {
    const int k0 = kt * BK;

    // ---- stage A tile (two 16B chunks per thread) ----
#if HAVE_ASYNC_LDS
    __builtin_amdgcn_global_load_async_to_lds_b128(
        (gv4i*)(A + (size_t)(m0 + arow) * K + k0 + acol),
        (lv4i*)&As[arow * AS + acol], 0, 0);
    __builtin_amdgcn_global_load_async_to_lds_b128(
        (gv4i*)(A + (size_t)(m0 + arow + 64) * K + k0 + acol),
        (lv4i*)&As[(arow + 64) * AS + acol], 0, 0);
#else
    {
      v8bf av0 = *(const v8bf*)(A + (size_t)(m0 + arow) * K + k0 + acol);
      v8bf av1 = *(const v8bf*)(A + (size_t)(m0 + arow + 64) * K + k0 + acol);
      *(v8bf*)(&As[arow * AS + acol]) = av0;
      *(v8bf*)(&As[(arow + 64) * AS + acol]) = av1;
    }
#endif

    // ---- stage B tile transposed (16B per thread, coalesced over n) ----
    {
      v8bf bv = *(const v8bf*)(Bw + (size_t)(k0 + bk) * N + n0 + bn);
#pragma unroll
      for (int j = 0; j < 8; ++j) Bt[(bn + j) * BS + bk] = bv[j];
    }

    if (kt + 1 < nK) {
      __builtin_prefetch(A + (size_t)(m0 + arow) * K + k0 + BK + acol, 0, 1);
      __builtin_prefetch(Bw + (size_t)(k0 + BK + bk) * N + n0 + bn, 0, 1);
    }
#if HAVE_ASYNC_LDS
    asm volatile("s_wait_asynccnt 0x0" ::: "memory");
#endif
    __syncthreads();

    // ---- fragments ----
    // A: 16-bit 16x32 layout. lanes 0-15: K 0-7 / 16-23; lanes 16-31: 8-15 / 24-31.
    const int ka = hi ? 8 : 0;
    // B: per-lane column, same K order as A; lanes 0-15: K 0-7 / 16-23; lanes 16-31: 8-15 / 24-31.
    const int kb = hi ? 8 : 0;
    union Frag { v16bf v; v8bf h[2]; };
    Frag a[2], b[2];
#pragma unroll
    for (int mi = 0; mi < 2; ++mi) {
      const int row = wm + mi * 16 + fr;
      a[mi].h[0] = *(const v8bf*)(&As[row * AS + ka]);
      a[mi].h[1] = *(const v8bf*)(&As[row * AS + 16 + ka]);
    }
#pragma unroll
    for (int ni = 0; ni < 2; ++ni) {
      const int col = wn + ni * 16 + fr;
      b[ni].h[0] = *(const v8bf*)(&Bt[col * BS + kb]);
      b[ni].h[1] = *(const v8bf*)(&Bt[col * BS + kb + 16]);
    }

#pragma unroll
    for (int mi = 0; mi < 2; ++mi)
#pragma unroll
      for (int ni = 0; ni < 2; ++ni)
        acc[mi][ni] = __builtin_amdgcn_wmma_f32_16x16x32_bf16(
            false, a[mi].v, false, b[ni].v, (short)0, acc[mi][ni], false, false);

    __syncthreads();
  }

  // ---- store: VGPR r -> M = tile_m + (hi?8:0) + r, N = tile_n + fr ----
#pragma unroll
  for (int mi = 0; mi < 2; ++mi) {
#pragma unroll
    for (int ni = 0; ni < 2; ++ni) {
      const int gn = n0 + wn + ni * 16 + fr;
      const int mb = m0 + wm + mi * 16 + hi * 8;
#pragma unroll
      for (int r = 0; r < 8; ++r) {
        const size_t idx = (size_t)(mb + r) * N + gn;
        float v = acc[mi][ni][r];
        if (residual) v += residual[idx];
        C[idx] = v;
      }
    }
  }
}

// --------------------- depthwise causal conv + SiLU ------------------------
__global__ __launch_bounds__(256)
void conv_silu_kernel(const float* __restrict__ xz, const float* __restrict__ cw,
                      const float* __restrict__ cb, float* __restrict__ u,
                      __bf16* __restrict__ u_bf) {
  const int idx = blockIdx.x * 256 + threadIdx.x;   // NTOK*D_INNER
  const int d = idx % D_INNER;
  const int t = idx / D_INNER;
  const int l = t % SEQ;
  float acc = cb[d];
#pragma unroll
  for (int j = 0; j < 4; ++j) {
    const int ls = l - 3 + j;
    if (ls >= 0)
      acc += cw[d * 4 + j] * xz[(size_t)(t - (3 - j)) * (2 * D_INNER) + d];
  }
  const float s = acc / (1.f + __expf(-acc));
  u[idx] = s;
  u_bf[idx] = (__bf16)s;
}

// --------------------------- small epilogues -------------------------------
__global__ __launch_bounds__(256)
void dtr_cast_kernel(const float* __restrict__ xdbl, __bf16* __restrict__ dtr) {
  int i = blockIdx.x * 256 + threadIdx.x;   // NTOK*DT_RANK
  int m = i / DT_RANK, j = i % DT_RANK;
  dtr[i] = (__bf16)xdbl[(size_t)m * XPROJ_N + j];
}

__global__ __launch_bounds__(256)
void softplus_kernel(float* __restrict__ dlt, const float* __restrict__ bdt) {
  int i = blockIdx.x * 256 + threadIdx.x;   // NTOK*D_INNER
  float v = dlt[i] + bdt[i % D_INNER];
  dlt[i] = (v > 20.f) ? v : log1pf(__expf(v));
}

// --------------------------- selective scan --------------------------------
// 32 blocks x 256 threads: thread owns channel (batch, d), h[64] in VGPRs,
// B/C broadcast through LDS each step. Fuses (y + u*D) * silu(z) -> bf16.
__global__ __launch_bounds__(256)
void scan_kernel(const float* __restrict__ delta, const float* __restrict__ u,
                 const float* __restrict__ xdbl, const float* __restrict__ aneg,
                 const float* __restrict__ Dp, const float* __restrict__ xz,
                 __bf16* __restrict__ yg) {
  __shared__ float Bs[D_STATE];
  __shared__ float Cs[D_STATE];
  const int tid = threadIdx.x;
  const int batch = blockIdx.x >> 3;
  const int d = (blockIdx.x & 7) * 256 + tid;

  float h[D_STATE];
  float Ar[D_STATE];
#pragma unroll
  for (int n = 0; n < D_STATE; ++n) {
    h[n] = 0.f;
    Ar[n] = aneg[(size_t)d * D_STATE + n];
  }
  const float Dv = Dp[d];

  for (int l = 0; l < SEQ; ++l) {
    const size_t tok = (size_t)batch * SEQ + l;
    if (tid < D_STATE)
      Bs[tid] = xdbl[tok * XPROJ_N + DT_RANK + tid];
    else if (tid < 2 * D_STATE)
      Cs[tid - D_STATE] = xdbl[tok * XPROJ_N + DT_RANK + D_STATE + (tid - D_STATE)];
    __syncthreads();

    const float dv = delta[tok * D_INNER + d];
    const float uv = u[tok * D_INNER + d];
    const float du = dv * uv;
    float y = 0.f;
#pragma unroll
    for (int n = 0; n < D_STATE; ++n) {
      h[n] = h[n] * __expf(dv * Ar[n]) + du * Bs[n];
      y += h[n] * Cs[n];
    }
    const float zv = xz[tok * (2 * D_INNER) + D_INNER + d];
    const float g = zv / (1.f + __expf(-zv));
    yg[tok * D_INNER + d] = (__bf16)((y + uv * Dv) * g);
    __syncthreads();
  }
}

// ------------------------------- launcher ----------------------------------
extern "C" void kernel_launch(void* const* d_in, const int* in_sizes, int n_in,
                              void* d_out, int out_size, void* d_ws, size_t ws_size,
                              hipStream_t stream) {
  (void)in_sizes; (void)n_in; (void)out_size; (void)ws_size;

  const float* x      = (const float*)d_in[0];
  const float* ln_w   = (const float*)d_in[1];
  const float* ln_b   = (const float*)d_in[2];
  const float* W_in   = (const float*)d_in[3];
  const float* conv_w = (const float*)d_in[4];
  const float* conv_b = (const float*)d_in[5];
  const float* W_xp   = (const float*)d_in[6];
  const float* W_dt   = (const float*)d_in[7];
  const float* b_dt   = (const float*)d_in[8];
  const float* A_log  = (const float*)d_in[9];
  const float* Dp     = (const float*)d_in[10];
  const float* W_out  = (const float*)d_in[11];
  float* out = (float*)d_out;

  char* ws = (char*)d_ws;
  size_t off = 0;
  auto alloc = [&](size_t bytes) -> void* {
    void* p = ws + off;
    off += (bytes + 255) & ~(size_t)255;
    return p;
  };

  __bf16* xn_bf   = (__bf16*)alloc((size_t)NTOK * D_MODEL * 2);
  __bf16* Win_bf  = (__bf16*)alloc((size_t)D_MODEL * 4096 * 2);
  __bf16* Wxp_bf  = (__bf16*)alloc((size_t)D_INNER * XPROJ_N * 2);
  __bf16* Wdt_bf  = (__bf16*)alloc((size_t)DT_RANK * D_INNER * 2);
  __bf16* Wout_bf = (__bf16*)alloc((size_t)D_INNER * D_MODEL * 2);
  float*  xz      = (float*)alloc((size_t)NTOK * 4096 * 4);
  float*  u       = (float*)alloc((size_t)NTOK * D_INNER * 4);
  __bf16* u_bf    = (__bf16*)alloc((size_t)NTOK * D_INNER * 2);
  float*  xdbl    = (float*)alloc((size_t)NTOK * XPROJ_N * 4);
  __bf16* dtr     = (__bf16*)alloc((size_t)NTOK * DT_RANK * 2);
  float*  dlt     = (float*)alloc((size_t)NTOK * D_INNER * 4);
  float*  aneg    = (float*)alloc((size_t)D_INNER * D_STATE * 4);
  __bf16* yg      = (__bf16*)alloc((size_t)NTOK * D_INNER * 2);

  dim3 blk(256);

  // weight preparation
  cast_bf16_kernel<<<(D_MODEL * 4096 + 255) / 256, blk, 0, stream>>>(W_in, Win_bf, D_MODEL * 4096);
  cast_bf16_kernel<<<(D_INNER * XPROJ_N + 255) / 256, blk, 0, stream>>>(W_xp, Wxp_bf, D_INNER * XPROJ_N);
  cast_bf16_kernel<<<(DT_RANK * D_INNER + 255) / 256, blk, 0, stream>>>(W_dt, Wdt_bf, DT_RANK * D_INNER);
  cast_bf16_kernel<<<(D_INNER * D_MODEL + 255) / 256, blk, 0, stream>>>(W_out, Wout_bf, D_INNER * D_MODEL);
  aneg_kernel<<<(D_INNER * D_STATE) / 256, blk, 0, stream>>>(A_log, aneg);

  // layernorm -> bf16 activations
  ln_kernel<<<NTOK, blk, 0, stream>>>(x, ln_w, ln_b, xn_bf);

  // xz = xn @ W_in          [8192,1024]x[1024,4096]
  gemm_bf16_wmma<<<dim3(4096 / BN, NTOK / BM), blk, 0, stream>>>(
      xn_bf, Win_bf, xz, nullptr, NTOK, 4096, D_MODEL);

  // u = silu(conv(xi) + b)
  conv_silu_kernel<<<(NTOK * D_INNER) / 256, blk, 0, stream>>>(xz, conv_w, conv_b, u, u_bf);

  // x_dbl = u @ W_xproj     [8192,2048]x[2048,192]
  gemm_bf16_wmma<<<dim3(XPROJ_N / BN, NTOK / BM), blk, 0, stream>>>(
      u_bf, Wxp_bf, xdbl, nullptr, NTOK, XPROJ_N, D_INNER);

  // delta = softplus(dt_r @ W_dt + b_dt)   [8192,64]x[64,2048]
  dtr_cast_kernel<<<(NTOK * DT_RANK) / 256, blk, 0, stream>>>(xdbl, dtr);
  gemm_bf16_wmma<<<dim3(D_INNER / BN, NTOK / BM), blk, 0, stream>>>(
      dtr, Wdt_bf, dlt, nullptr, NTOK, D_INNER, DT_RANK);
  softplus_kernel<<<(NTOK * D_INNER) / 256, blk, 0, stream>>>(dlt, b_dt);

  // selective scan + gating -> bf16
  scan_kernel<<<NB * 8, blk, 0, stream>>>(dlt, u, xdbl, aneg, Dp, xz, yg);

  // out = residual + yg @ W_out   [8192,2048]x[2048,1024]
  gemm_bf16_wmma<<<dim3(D_MODEL / BN, NTOK / BM), blk, 0, stream>>>(
      yg, Wout_bf, out, x, NTOK, D_MODEL, D_INNER);
}
